// Block_30709016166763
// MI455X (gfx1250) — matrix operand order repaired and checked by reading, not compiled, not run on hardware
//
#include <hip/hip_runtime.h>
#include <hip/hip_bf16.h>

// ---------------- types ----------------
typedef __attribute__((ext_vector_type(16))) __bf16          v16bf;
typedef __attribute__((ext_vector_type(16))) unsigned short  v16u;
typedef __attribute__((ext_vector_type(8)))  unsigned short  ushort8;
typedef __attribute__((ext_vector_type(8)))  float           v8f;

#define BB   2
#define TT   2048
#define EE   1024
#define HH   16
#define HS   64
#define FF   4096
#define ROWS (BB * TT)          // 4096
#define LDQKV (3 * EE)          // 3072

static __device__ __forceinline__ unsigned short f2bf(float f) {
    unsigned int u = __builtin_bit_cast(unsigned int, f);
    u += 0x7fffu + ((u >> 16) & 1u);          // round-to-nearest-even
    return (unsigned short)(u >> 16);
}

// Build a 16-element bf16 fragment from two contiguous 8-element chunks.
static __device__ __forceinline__ v16bf frag16(const unsigned short* p0,
                                               const unsigned short* p1) {
    ushort8 a = *(const ushort8*)p0;
    ushort8 b = *(const ushort8*)p1;
    v16u u;
#pragma unroll
    for (int i = 0; i < 8; ++i) { u[i] = a[i]; u[i + 8] = b[i]; }
    return __builtin_bit_cast(v16bf, u);
}

static __device__ __forceinline__ v8f wmma_bf16(v16bf a, v16bf b, v8f c) {
    return __builtin_amdgcn_wmma_f32_16x16x32_bf16(false, a, false, b,
                                                   (short)0, c, false, false);
}

// ---------------- weight conversion ----------------
__global__ __launch_bounds__(256) void f2bf_kernel(const float* __restrict__ in,
                                                   unsigned short* __restrict__ out,
                                                   int n) {
    int i = blockIdx.x * 256 + threadIdx.x;
    if (i < n) out[i] = f2bf(in[i]);
}

// Wqkv[e][j], j in [0,3072): sel=j/1024 picks q/k/v, h=(j%1024)/64, d=j%64
__global__ __launch_bounds__(256) void build_wqkv_kernel(const float* __restrict__ wq,
                                                         const float* __restrict__ wk,
                                                         const float* __restrict__ wv,
                                                         unsigned short* __restrict__ out) {
    int idx = blockIdx.x * 256 + threadIdx.x;
    if (idx >= EE * LDQKV) return;
    int e = idx / LDQKV, j = idx % LDQKV;
    int sel = j >> 10, jj = j & 1023;
    int h = jj >> 6, d = jj & 63;
    const float* src = (sel == 0) ? wq : (sel == 1) ? wk : wv;
    out[idx] = f2bf(src[((size_t)h * EE + e) * HS + d]);
}

// ---------------- layernorm (row of 1024) -> bf16 ----------------
__global__ __launch_bounds__(256) void ln_bf16_kernel(const float* __restrict__ x,
                                                      const float* __restrict__ g,
                                                      const float* __restrict__ be,
                                                      unsigned short* __restrict__ out) {
    int row = blockIdx.x;
    const float* xr = x + (size_t)row * EE;
    float v[4], s = 0.f, s2 = 0.f;
#pragma unroll
    for (int i = 0; i < 4; ++i) {
        v[i] = xr[threadIdx.x + i * 256];
        s += v[i]; s2 += v[i] * v[i];
    }
#pragma unroll
    for (int m = 16; m >= 1; m >>= 1) {
        s  += __shfl_xor(s,  m, 32);
        s2 += __shfl_xor(s2, m, 32);
    }
    __shared__ float red[16];
    int wid = threadIdx.x >> 5, lane = threadIdx.x & 31;
    if (lane == 0) { red[wid] = s; red[8 + wid] = s2; }
    __syncthreads();
    s = 0.f; s2 = 0.f;
#pragma unroll
    for (int k = 0; k < 8; ++k) { s += red[k]; s2 += red[8 + k]; }
    float mu  = s * (1.f / EE);
    float var = s2 * (1.f / EE) - mu * mu;
    float rs  = rsqrtf(var + 1e-5f);
    unsigned short* orow = out + (size_t)row * EE;
#pragma unroll
    for (int i = 0; i < 4; ++i) {
        int c = threadIdx.x + i * 256;
        orow[c] = f2bf((v[i] - mu) * rs * g[c] + be[c]);
    }
}

// ---------------- generic bf16 WMMA GEMM (flags specialized at compile time) ----
// C[M,N] = A[M,K](bf16) * B[K,N](bf16) [+bias][+relu][+resid] -> outF and/or outB
// M,N multiples of 128; K multiple of 32. Software-pipelined global->LDS staging.
template <bool HAS_BIAS, bool HAS_RES, bool DO_RELU, bool WR_F32, bool WR_BF16>
__global__ __launch_bounds__(256) void gemm_bf16_kernel(
        const unsigned short* __restrict__ A,
        const unsigned short* __restrict__ Bw,
        float* __restrict__ outF, unsigned short* __restrict__ outB,
        const float* __restrict__ bias, const float* __restrict__ resid,
        int M, int N, int K) {
    __shared__ unsigned short As[128][40];   // rows padded: 80B stride, 16B aligned
    __shared__ unsigned short Bt[128][40];   // B transposed: [n][k]
    int tid = threadIdx.x, lane = tid & 31, wid = tid >> 5;
    int wM = wid >> 1, wN = wid & 1;
    int bM = blockIdx.y * 128, bN = blockIdx.x * 128;

    // per-thread staging assignments
    int arow = tid >> 1, acol = (tid & 1) * 16;      // A: 128 rows x 2 segs
    int brow = tid >> 3, bcol = (tid & 7) * 16;      // B: 32 rows x 8 segs
    const unsigned short* Ag = A + (size_t)(bM + arow) * K + acol;
    const unsigned short* Bg = Bw + (size_t)brow * N + bN + bcol;

    // prefetch tile 0 into registers
    ushort8 ra0 = *(const ushort8*)Ag;
    ushort8 ra1 = *(const ushort8*)(Ag + 8);
    ushort8 rb0 = *(const ushort8*)Bg;
    ushort8 rb1 = *(const ushort8*)(Bg + 8);

    v8f acc[2][4] = {};

    int r    = lane & 15;
    int koff = (lane >> 4) << 3;   // A layout half select (0 / 8)
    int kh   = (lane >> 4) << 4;   // unused; B also selects k halves with koff (0 / 8)

    for (int k0 = 0; k0 < K; k0 += 32) {
        __syncthreads();           // previous compute finished reading LDS
        *(ushort8*)&As[arow][acol]     = ra0;
        *(ushort8*)&As[arow][acol + 8] = ra1;
#pragma unroll
        for (int i = 0; i < 8; ++i) {
            Bt[bcol + i][brow]     = rb0[i];
            Bt[bcol + 8 + i][brow] = rb1[i];
        }
        __syncthreads();

        // prefetch next tile while this chunk computes (loads in flight over WMMAs)
        if (k0 + 32 < K) {
            const unsigned short* An = Ag + k0 + 32;
            const unsigned short* Bn = Bg + (size_t)(k0 + 32) * N;
            ra0 = *(const ushort8*)An;
            ra1 = *(const ushort8*)(An + 8);
            rb0 = *(const ushort8*)Bn;
            rb1 = *(const ushort8*)(Bn + 8);
        }

        const unsigned short* a0p = &As[wM * 32 + r][0];
        const unsigned short* a1p = &As[wM * 32 + 16 + r][0];
        v16bf a0 = frag16(a0p + koff, a0p + 16 + koff);
        v16bf a1 = frag16(a1p + koff, a1p + 16 + koff);
#pragma unroll
        for (int j = 0; j < 4; ++j) {
            const unsigned short* bp = &Bt[wN * 64 + j * 16 + r][koff];
            v16bf bf = frag16(bp, bp + 16);
            acc[0][j] = wmma_bf16(a0, bf, acc[0][j]);
            acc[1][j] = wmma_bf16(a1, bf, acc[1][j]);
        }
    }

    // epilogue: C layout — N = lane&15, M = e + 8*(lane>>4). Fully specialized.
    int mh = (lane >> 4) << 3;
#pragma unroll
    for (int i = 0; i < 2; ++i) {
#pragma unroll
        for (int j = 0; j < 4; ++j) {
            int col = bN + wN * 64 + j * 16 + r;
            float bv = HAS_BIAS ? bias[col] : 0.f;
#pragma unroll
            for (int e = 0; e < 8; ++e) {
                int row = bM + wM * 32 + i * 16 + mh + e;
                float vv = acc[i][j][e] + bv;
                if (DO_RELU) vv = fmaxf(vv, 0.f);
                size_t off = (size_t)row * N + col;
                if (HAS_RES) vv += resid[off];
                if (WR_F32)  outF[off] = vv;
                if (WR_BF16) outB[off] = f2bf(vv);
            }
        }
    }
}

// ---------------- flash attention (causal, scale 1/sqrt(E)=1/32) ----------------
// grid: (T/64, B*H), block 128 (4 waves x 16 q-rows). qkv: [ROWS][3E] bf16.
__global__ __launch_bounds__(128) void attn_kernel(const unsigned short* __restrict__ qkv,
                                                   unsigned short* __restrict__ attn) {
    __shared__ unsigned short Ks[32][72];       // [key][d], rows 144B
    __shared__ unsigned short Vt[64][40];       // [d][key], rows 80B
    __shared__ unsigned short Pl[4][16][40];    // per-wave P tile [q][key]

    int qt = blockIdx.x, bh = blockIdx.y;
    int b = bh >> 4, h = bh & 15;
    int tid = threadIdx.x, lane = tid & 31, w = tid >> 5;
    int r    = lane & 15;
    int koff = (lane >> 4) << 3;
    int kh   = (lane >> 4) << 4;

    const unsigned short* qb =
        qkv + (size_t)(b * TT + qt * 64 + w * 16) * LDQKV + h * HS;
    const unsigned short* qr = qb + (size_t)r * LDQKV;
    v16bf qf0 = frag16(qr + koff,      qr + 16 + koff);   // d 0..31
    v16bf qf1 = frag16(qr + 32 + koff, qr + 48 + koff);   // d 32..63

    float runmax[8], runsum[8];
    v8f   o[4] = {};
#pragma unroll
    for (int e = 0; e < 8; ++e) { runmax[e] = -1e30f; runsum[e] = 0.f; }
    int qrow0 = qt * 64 + w * 16 + ((lane >> 4) << 3);   // + e gives this lane's q row

    int nkb = 2 * (qt + 1);
    for (int kb = 0; kb < nkb; ++kb) {
        int s0 = kb * 32;
        __syncthreads();
        {   // stage K (row-major) and V (transposed) tiles: 32 keys x 64 d
            int sr = tid >> 2, seg = (tid & 3) * 16;
            const unsigned short* kg =
                qkv + (size_t)(b * TT + s0 + sr) * LDQKV + EE + h * HS + seg;
            *(ushort8*)&Ks[sr][seg]     = *(const ushort8*)kg;
            *(ushort8*)&Ks[sr][seg + 8] = *(const ushort8*)(kg + 8);
            const unsigned short* vg =
                qkv + (size_t)(b * TT + s0 + sr) * LDQKV + 2 * EE + h * HS + seg;
            ushort8 v0 = *(const ushort8*)vg;
            ushort8 v1 = *(const ushort8*)(vg + 8);
#pragma unroll
            for (int i = 0; i < 8; ++i) {
                Vt[seg + i][sr]     = v0[i];
                Vt[seg + 8 + i][sr] = v1[i];
            }
        }
        __syncthreads();

        // ---- scores: two 16x16 tiles, contraction over d=64 (2 WMMAs each)
        float sf[2][8];
#pragma unroll
        for (int st = 0; st < 2; ++st) {
            v8f sa = {};
            const unsigned short* kp = &Ks[st * 16 + r][koff];
            v16bf b0 = frag16(kp,      kp + 16);      // d 0..31
            v16bf b1 = frag16(kp + 32, kp + 48);      // d 32..63
            sa = wmma_bf16(qf0, b0, sa);
            sa = wmma_bf16(qf1, b1, sa);
            int sidx = s0 + st * 16 + r;              // key col for this lane
#pragma unroll
            for (int e = 0; e < 8; ++e) {
                float sv = sa[e] * 0.03125f;
                if (sidx > qrow0 + e) sv = -1e30f;    // causal mask
                sf[st][e] = sv;
            }
        }
        // ---- online softmax update
        float tmax[8];
#pragma unroll
        for (int e = 0; e < 8; ++e) tmax[e] = fmaxf(sf[0][e], sf[1][e]);
#pragma unroll
        for (int m = 8; m >= 1; m >>= 1)
#pragma unroll
            for (int e = 0; e < 8; ++e)
                tmax[e] = fmaxf(tmax[e], __shfl_xor(tmax[e], m, 32));
        float corr[8], p0[8], p1[8], psum[8];
#pragma unroll
        for (int e = 0; e < 8; ++e) {
            float nm = fmaxf(runmax[e], tmax[e]);
            corr[e] = __expf(runmax[e] - nm);
            runmax[e] = nm;
            p0[e] = __expf(sf[0][e] - nm);
            p1[e] = __expf(sf[1][e] - nm);
            psum[e] = p0[e] + p1[e];
        }
#pragma unroll
        for (int m = 8; m >= 1; m >>= 1)
#pragma unroll
            for (int e = 0; e < 8; ++e)
                psum[e] += __shfl_xor(psum[e], m, 32);
#pragma unroll
        for (int e = 0; e < 8; ++e)
            runsum[e] = runsum[e] * corr[e] + psum[e];
#pragma unroll
        for (int j = 0; j < 4; ++j)
#pragma unroll
            for (int e = 0; e < 8; ++e)
                o[j][e] = o[j][e] * corr[e];

        // ---- P -> per-wave LDS (C layout -> row-major), then P@V
        int prow = (lane >> 4) << 3;
#pragma unroll
        for (int e = 0; e < 8; ++e) {
            Pl[w][prow + e][r]      = f2bf(p0[e]);
            Pl[w][prow + e][16 + r] = f2bf(p1[e]);
        }
        const unsigned short* pp = &Pl[w][r][0];      // same-wave DS ops are in-order
        v16bf pf = frag16(pp + koff, pp + 16 + koff);
#pragma unroll
        for (int j = 0; j < 4; ++j) {
            const unsigned short* vp = &Vt[j * 16 + r][koff];
            v16bf vf = frag16(vp, vp + 16);
            o[j] = wmma_bf16(pf, vf, o[j]);
        }
    }

    // ---- normalize and store (bf16, concat-head layout [ROWS][E])
    unsigned short* ob =
        attn + (size_t)(b * TT + qt * 64 + w * 16) * EE + h * HS;
    int mh = (lane >> 4) << 3;
#pragma unroll
    for (int j = 0; j < 4; ++j)
#pragma unroll
        for (int e = 0; e < 8; ++e)
            ob[(size_t)(mh + e) * EE + j * 16 + r] = f2bf(o[j][e] / runsum[e]);
}

// ---------------- launch ----------------
extern "C" void kernel_launch(void* const* d_in, const int* in_sizes, int n_in,
                              void* d_out, int out_size, void* d_ws, size_t ws_size,
                              hipStream_t stream) {
    const float* x      = (const float*)d_in[0];
    const float* wq     = (const float*)d_in[1];
    const float* wk     = (const float*)d_in[2];
    const float* wv     = (const float*)d_in[3];
    const float* w_proj = (const float*)d_in[4];
    const float* b_proj = (const float*)d_in[5];
    const float* w1     = (const float*)d_in[6];
    const float* b1     = (const float*)d_in[7];
    const float* w2     = (const float*)d_in[8];
    const float* b2     = (const float*)d_in[9];
    const float* g1     = (const float*)d_in[10];
    const float* be1    = (const float*)d_in[11];
    const float* g2     = (const float*)d_in[12];
    const float* be2    = (const float*)d_in[13];
    float* out = (float*)d_out;

    // scratch partition (~120 MB)
    unsigned short* qkvw   = (unsigned short*)d_ws;                 // [E][3E]
    unsigned short* wprojb = qkvw   + (size_t)EE * LDQKV;           // [E][E]
    unsigned short* w1b    = wprojb + (size_t)EE * EE;              // [E][4E]
    unsigned short* w2b    = w1b    + (size_t)EE * FF;              // [4E][E]
    unsigned short* hln    = w2b    + (size_t)FF * EE;              // [ROWS][E]
    unsigned short* qkv    = hln    + (size_t)ROWS * EE;            // [ROWS][3E]
    unsigned short* attnb  = qkv    + (size_t)ROWS * LDQKV;         // [ROWS][E]
    float*          x1     = (float*)(attnb + (size_t)ROWS * EE);   // [ROWS][E]
    unsigned short* hln2   = (unsigned short*)(x1 + (size_t)ROWS * EE);
    unsigned short* ff1    = hln2 + (size_t)ROWS * EE;              // [ROWS][4E]

    // 1) weights -> bf16
    build_wqkv_kernel<<<(EE * LDQKV + 255) / 256, 256, 0, stream>>>(wq, wk, wv, qkvw);
    f2bf_kernel<<<(EE * EE + 255) / 256, 256, 0, stream>>>(w_proj, wprojb, EE * EE);
    f2bf_kernel<<<(EE * FF + 255) / 256, 256, 0, stream>>>(w1, w1b, EE * FF);
    f2bf_kernel<<<(FF * EE + 255) / 256, 256, 0, stream>>>(w2, w2b, FF * EE);

    // 2) LN1
    ln_bf16_kernel<<<ROWS, 256, 0, stream>>>(x, g1, be1, hln);

    // 3) fused QKV GEMM: [4096,1024] x [1024,3072] -> bf16
    gemm_bf16_kernel<false, false, false, false, true>
        <<<dim3(LDQKV / 128, ROWS / 128), 256, 0, stream>>>(
            hln, qkvw, nullptr, qkv, nullptr, nullptr, ROWS, LDQKV, EE);

    // 4) causal flash attention
    attn_kernel<<<dim3(TT / 64, BB * HH), 128, 0, stream>>>(qkv, attnb);

    // 5) output projection + bias + residual(x) -> x1 (fp32)
    gemm_bf16_kernel<true, true, false, true, false>
        <<<dim3(EE / 128, ROWS / 128), 256, 0, stream>>>(
            attnb, wprojb, x1, nullptr, b_proj, x, ROWS, EE, EE);

    // 6) LN2
    ln_bf16_kernel<<<ROWS, 256, 0, stream>>>(x1, g2, be2, hln2);

    // 7) FFN up + bias + relu -> bf16
    gemm_bf16_kernel<true, false, true, false, true>
        <<<dim3(FF / 128, ROWS / 128), 256, 0, stream>>>(
            hln2, w1b, nullptr, ff1, b1, nullptr, ROWS, FF, EE);

    // 8) FFN down + bias + residual(x1) -> out (fp32)
    gemm_bf16_kernel<true, true, false, true, false>
        <<<dim3(EE / 128, ROWS / 128), 256, 0, stream>>>(
            ff1, w2b, out, nullptr, b2, x1, ROWS, EE, FF);
}
